// QuanvolutionGraphQLClassifier_65481071409985
// MI455X (gfx1250) — matrix operand order repaired and checked by reading, not compiled, not run on hardware
//
#include <hip/hip_runtime.h>
#include <math.h>

#define BB 4096
#define NP 196
#define KP 800    // K = 784 padded to 800 (25 * 32)
#define NSTEP 25  // K-steps of 32 for wmma_f32_16x16x32_f16

typedef __attribute__((ext_vector_type(16))) _Float16 v16h;
typedef __attribute__((ext_vector_type(8)))  _Float16 v8h;
typedef __attribute__((ext_vector_type(4)))  _Float16 v4h;
typedef __attribute__((ext_vector_type(8)))  float    v8f;

// ---------------- 4-qubit gate helpers (state idx = w0*8 + w1*4 + w2*2 + w3) ----

template<int W>
__device__ __forceinline__ void ry_gate(float* re, float* im, float c, float s) {
  const int st = 8 >> W;
  #pragma unroll
  for (int i = 0; i < 16; ++i) {
    if (!(i & st)) {
      const int j = i | st;
      float r0 = re[i], r1 = re[j], i0 = im[i], i1 = im[j];
      re[i] = c * r0 - s * r1;  im[i] = c * i0 - s * i1;
      re[j] = s * r0 + c * r1;  im[j] = s * i0 + c * i1;
    }
  }
}

template<int W>
__device__ __forceinline__ void rx_gate(float* re, float* im, float c, float s) {
  const int st = 8 >> W;
  #pragma unroll
  for (int i = 0; i < 16; ++i) {
    if (!(i & st)) {
      const int j = i | st;
      float r0 = re[i], r1 = re[j], i0 = im[i], i1 = im[j];
      // |0>' = c|0> - i s|1> ;  |1>' = -i s|0> + c|1>
      re[i] = c * r0 + s * i1;  im[i] = c * i0 - s * r1;
      re[j] = c * r1 + s * i0;  im[j] = c * i1 - s * r0;
    }
  }
}

template<int W>
__device__ __forceinline__ void rz_gate(float* re, float* im, float c, float s) {
  const int st = 8 >> W;
  #pragma unroll
  for (int i = 0; i < 16; ++i) {
    if (!(i & st)) {
      const int j = i | st;
      float r0 = re[i], r1 = re[j], i0 = im[i], i1 = im[j];
      // |0> *= (c - i s) ; |1> *= (c + i s)
      re[i] = c * r0 + s * i0;  im[i] = c * i0 - s * r0;
      re[j] = c * r1 - s * i1;  im[j] = c * i1 + s * r1;
    }
  }
}

template<int C, int T>
__device__ __forceinline__ void cnot_gate(float* re, float* im) {
  const int sc = 8 >> C, st = 8 >> T;
  #pragma unroll
  for (int i = 0; i < 16; ++i) {
    if ((i & sc) && !(i & st)) {
      const int j = i | st;
      float t;
      t = re[i]; re[i] = re[j]; re[j] = t;
      t = im[i]; im[i] = im[j]; im[j] = t;
    }
  }
}

// ---------------- Kernel 1: quantum patch measurement ---------------------------
// One thread per patch. State init |0000> + RY(angle) per wire == product state.
// Writes meas as f16 rows of length KP=800 (zero-padded), plus sample-0 in fp32.

__global__ void __launch_bounds__(256)
quanv_kernel(const float* __restrict__ x, const float* __restrict__ qp,
             _Float16* __restrict__ measH, float* __restrict__ meas0) {
  int pid = blockIdx.x * blockDim.x + threadIdx.x;
  if (pid >= BB * NP) return;
  int b  = pid / NP;
  int pp = pid - b * NP;
  int pi = pp / 14;
  int pj = pp - pi * 14;

  const float* xb = x + (size_t)b * 784 + pi * 56 + pj * 2;
  float a0 = xb[0], a1 = xb[1], a2 = xb[28], a3 = xb[29]; // tl,tr,bl,br -> wires 0..3

  float c0, s0, c1, s1, c2, s2, c3, s3;
  sincosf(a0 * 0.5f, &s0, &c0);
  sincosf(a1 * 0.5f, &s1, &c1);
  sincosf(a2 * 0.5f, &s2, &c2);
  sincosf(a3 * 0.5f, &s3, &c3);

  float qc[6], qs[6];
  #pragma unroll
  for (int k = 0; k < 6; ++k) sincosf(qp[k] * 0.5f, &qs[k], &qc[k]);

  float re[16], im[16];
  #pragma unroll
  for (int i = 0; i < 16; ++i) {
    re[i] = ((i & 8) ? s0 : c0) * ((i & 4) ? s1 : c1) *
            ((i & 2) ? s2 : c2) * ((i & 1) ? s3 : c3);
    im[i] = 0.0f;
  }

  rx_gate<0>(re, im, qc[0], qs[0]);
  ry_gate<1>(re, im, qc[1], qs[1]);
  rz_gate<2>(re, im, qc[2], qs[2]);
  cnot_gate<0, 1>(re, im);
  rx_gate<3>(re, im, qc[3], qs[3]);
  cnot_gate<2, 3>(re, im);
  ry_gate<0>(re, im, qc[4], qs[4]);
  rz_gate<3>(re, im, qc[5], qs[5]);

  float e0 = 0.f, e1 = 0.f, e2 = 0.f, e3 = 0.f;
  #pragma unroll
  for (int i = 0; i < 16; ++i) {
    float p = re[i] * re[i] + im[i] * im[i];
    e0 += (i & 8) ? -p : p;
    e1 += (i & 4) ? -p : p;
    e2 += (i & 2) ? -p : p;
    e3 += (i & 1) ? -p : p;
  }

  v4h h;
  h[0] = (_Float16)e0; h[1] = (_Float16)e1; h[2] = (_Float16)e2; h[3] = (_Float16)e3;
  *(v4h*)(measH + (size_t)b * KP + pp * 4) = h;

  if (pp < 4) {  // zero the K-pad 784..799 of this row (16 halves, 4 threads x 4)
    v4h z = {};
    *(v4h*)(measH + (size_t)b * KP + 784 + pp * 4) = z;
  }
  if (b == 0) {
    meas0[pp * 4 + 0] = e0; meas0[pp * 4 + 1] = e1;
    meas0[pp * 4 + 2] = e2; meas0[pp * 4 + 3] = e3;
  }
}

// ---------------- Kernel 2a: adjacency from sample-0 cosine similarity ----------

__global__ void __launch_bounds__(256)
adj_kernel(const float* __restrict__ meas0, float* __restrict__ adj) {
  __shared__ float v[NP][4];
  __shared__ float nrm[NP];
  int t = threadIdx.x;
  if (t < NP) {
    float a = meas0[t * 4], b = meas0[t * 4 + 1];
    float c = meas0[t * 4 + 2], d = meas0[t * 4 + 3];
    v[t][0] = a; v[t][1] = b; v[t][2] = c; v[t][3] = d;
    nrm[t] = sqrtf(a * a + b * b + c * c + d * d);
  }
  __syncthreads();
  if (t < NP) {
    float a = v[t][0], b = v[t][1], c = v[t][2], d = v[t][3];
    float nt = nrm[t];
    for (int m = 0; m < NP; ++m) {
      float dot = a * v[m][0] + b * v[m][1] + c * v[m][2] + d * v[m][3];
      float sim = dot / (nt * nrm[m] + 1e-12f);
      adj[t * NP + m] = (sim >= 0.9f) ? 1.0f : ((sim >= 0.5f) ? 0.5f : 0.0f);
    }
  }
}

// ---------------- Kernel 2b: fold adjacency into classifier weights ------------
// W2[c, 4m+d] = sum_n W[c, 4n+d] * adj[n, m], emitted directly in the per-lane
// WMMA B-matrix layout: W2B[s][lane][h], K = s*32 + (lane>=16 ? 16 : 0) + h,
// col = lane & 15 (cols 10..15 and K >= 784 are zero).

__global__ void __launch_bounds__(256)
fold_kernel(const float* __restrict__ W, const float* __restrict__ adj,
            _Float16* __restrict__ W2B) {
  int tid = blockIdx.x * blockDim.x + threadIdx.x;
  if (tid >= NSTEP * 512) return;
  int h = tid & 15;
  int l = (tid >> 4) & 31;
  int s = tid >> 9;
  int col = l & 15;
  int K = s * 32 + ((l >> 4) << 4) + h;
  float acc = 0.0f;
  if (col < 10 && K < 784) {
    int d = K & 3;
    int m = K >> 2;
    const float* wrow = W + col * 784 + d;  // stride 4
    const float* acol = adj + m;            // stride NP
    for (int n = 0; n < NP; ++n) acc += wrow[4 * n] * acol[n * NP];
  }
  W2B[tid] = (_Float16)acc;
}

// ---------------- Kernel 3: WMMA GEMM (4096x784 @ 784x16) + log_softmax --------
// One wave per 16-row tile: 32 blocks x 8 waves. A layout (contiguous k, same order as B):
// lane L holds row L&15, K = (L>=16?16:0) + {0..15} -> two b128 loads.

__global__ void __launch_bounds__(256)
gemm_softmax_kernel(const _Float16* __restrict__ A, const _Float16* __restrict__ Bm,
                    const float* __restrict__ bias, float* __restrict__ out) {
  __shared__ float tile[8][16][17];
  int lane = threadIdx.x & 31;
  int w    = threadIdx.x >> 5;
  int rowBase = (blockIdx.x * 8 + w) * 16;

  const _Float16* aP = A + (size_t)(rowBase + (lane & 15)) * KP + ((lane >> 4) << 4);
  const _Float16* bP = Bm + lane * 16;

  v8f acc = {};
  for (int s = 0; s < NSTEP; ++s) {
    v8h alo = *(const v8h*)(aP + s * 32);        // K = s*32 + base + 0..7
    v8h ahi = *(const v8h*)(aP + s * 32 + 8);    // K = s*32 + base + 8..15
    v8h blo = *(const v8h*)(bP + s * 512);       // K = s*32 + (lane>=16?16:0) + 0..7
    v8h bhi = *(const v8h*)(bP + s * 512 + 8);   //                            + 8..15
    v16h Av = __builtin_shufflevector(alo, ahi, 0,1,2,3,4,5,6,7,8,9,10,11,12,13,14,15);
    v16h Bv = __builtin_shufflevector(blo, bhi, 0,1,2,3,4,5,6,7,8,9,10,11,12,13,14,15);
    acc = __builtin_amdgcn_wmma_f32_16x16x32_f16(false, Av, false, Bv,
                                                 (short)0, acc, false, false);
  }

  int col = lane & 15;
  float bc = (col < 10) ? bias[col] : 0.0f;
  int mhalf = (lane >> 4) << 3;
  #pragma unroll
  for (int r = 0; r < 8; ++r) tile[w][mhalf + r][col] = acc[r] + bc;
  __syncthreads();

  if (lane < 16) {
    const float* L = tile[w][lane];
    float mx = L[0];
    #pragma unroll
    for (int c = 1; c < 10; ++c) mx = fmaxf(mx, L[c]);
    float sum = 0.0f;
    #pragma unroll
    for (int c = 0; c < 10; ++c) sum += expf(L[c] - mx);
    float lse = mx + logf(sum);
    float* o = out + (size_t)(rowBase + lane) * 10;
    #pragma unroll
    for (int c = 0; c < 10; ++c) o[c] = L[c] - lse;
  }
}

// ---------------- Launch ---------------------------------------------------------

extern "C" void kernel_launch(void* const* d_in, const int* in_sizes, int n_in,
                              void* d_out, int out_size, void* d_ws, size_t ws_size,
                              hipStream_t stream) {
  const float* x  = (const float*)d_in[0];   // (4096,1,28,28)
  const float* qp = (const float*)d_in[1];   // (6,)
  const float* W  = (const float*)d_in[2];   // (10, 784)
  const float* bv = (const float*)d_in[3];   // (10,)
  float* out = (float*)d_out;                // (4096, 10)

  char* ws = (char*)d_ws;
  size_t off = 0;
  _Float16* measH = (_Float16*)(ws + off);
  off += (size_t)BB * KP * sizeof(_Float16);        // 6,553,600 B
  off = (off + 255) & ~(size_t)255;
  float* meas0 = (float*)(ws + off);
  off += (size_t)NP * 4 * sizeof(float);            // 3,136 B
  off = (off + 255) & ~(size_t)255;
  float* adj = (float*)(ws + off);
  off += (size_t)NP * NP * sizeof(float);           // 153,664 B
  off = (off + 255) & ~(size_t)255;
  _Float16* W2B = (_Float16*)(ws + off);            // 25,600 B

  int P = BB * NP;
  quanv_kernel<<<(P + 255) / 256, 256, 0, stream>>>(x, qp, measH, meas0);
  adj_kernel<<<1, 256, 0, stream>>>(meas0, adj);
  fold_kernel<<<(NSTEP * 512 + 255) / 256, 256, 0, stream>>>(W, adj, W2B);
  gemm_softmax_kernel<<<BB / 128, 256, 0, stream>>>(measH, W2B, bv, out);
}
